// AttentionBlock_54382875902273
// MI455X (gfx1250) — matrix operand order repaired and checked by reading, not compiled, not run on hardware
//
#include <hip/hip_runtime.h>
#include <hip/hip_bf16.h>
#include <stdint.h>

// Problem constants (match reference)
#define BATCH 4
#define SEQ   2048
#define HID   512
#define NHEAD 8
#define HDIM  64
#define MROWS (BATCH * SEQ)   // 8192 total rows for the projections

typedef __bf16 v16bf __attribute__((ext_vector_type(16)));
typedef __bf16 v8bf  __attribute__((ext_vector_type(8)));
typedef float  v8f   __attribute__((ext_vector_type(8)));

static __device__ __forceinline__ __bf16 f32_to_bf16(float f) {
  union { float f; uint32_t u; } a; a.f = f;
  uint32_t u = a.u;
  uint32_t r = (u + 0x7FFFu + ((u >> 16) & 1u)) >> 16;   // round-to-nearest-even
  union { uint16_t s; __bf16 b; } o; o.s = (uint16_t)r;
  return o.b;
}

static __device__ __forceinline__ v16bf cat8(v8bf lo, v8bf hi) {
  union { v16bf v; v8bf h[2]; } u;
  u.h[0] = lo; u.h[1] = hi;
  return u.v;
}

static __device__ __forceinline__ v8f wmma_bf16(v16bf a, v16bf b, v8f c) {
  // D = A(16x32 bf16) * B(32x16 bf16) + C(16x16 f32)
  return __builtin_amdgcn_wmma_f32_16x16x32_bf16(false, a, false, b, (short)0, c,
                                                 false, false);
}

// ---------------------------------------------------------------------------
// f32 -> bf16 conversion (grid-stride)
// ---------------------------------------------------------------------------
__global__ void cvt_f32_bf16(const float* __restrict__ in, __bf16* __restrict__ out,
                             int count) {
  int i = blockIdx.x * blockDim.x + threadIdx.x;
  int stride = gridDim.x * blockDim.x;
  for (; i < count; i += stride) out[i] = f32_to_bf16(in[i]);
}

// ---------------------------------------------------------------------------
// GEMM: out = X(M x 512) @ W^T(512 x 512, W is torch-style [out,in]) + bias
// mode 0: bf16 out, layout [B, NH, S, HD]          (Q, K projections)
// mode 1: bf16 out, layout [B, NH, HD, S] (transp.) (V projection)
// mode 2: f32  out, layout [M, 512]                 (final output projection)
// Block = 128 threads (4 waves). Block tile 64x64; wave tile 16x64.
// ---------------------------------------------------------------------------
__global__ __launch_bounds__(128) void gemm_xwT_kernel(
    const __bf16* __restrict__ X, const __bf16* __restrict__ W,
    const float* __restrict__ bias, void* __restrict__ outp, int mode) {
  const int wave  = threadIdx.x >> 5;
  const int lane  = threadIdx.x & 31;
  const int m0    = blockIdx.x * 64 + wave * 16;
  const int n0    = blockIdx.y * 64;
  const int row   = lane & 15;               // A-frag row
  const int kbase = (lane < 16) ? 0 : 8;     // unused; A-frag K base is khalf
  const int n     = lane & 15;               // B-frag column
  const int khalf = (lane < 16) ? 0 : 16;    // B-frag K half
  const int mbase = (lane < 16) ? 0 : 8;     // C-frag row base

  v8f acc[4] = {{}, {}, {}, {}};
  const __bf16* xr = X + (size_t)(m0 + row) * HID;

#pragma unroll 4
  for (int k = 0; k < HID; k += 32) {
    v16bf a = *(const v16bf*)(xr + k + khalf);
#pragma unroll
    for (int t = 0; t < 4; ++t) {
      const __bf16* wr = W + (size_t)(n0 + t * 16 + n) * HID + k + khalf;
      acc[t] = wmma_bf16(a, *(const v16bf*)wr, acc[t]);
    }
  }

#pragma unroll
  for (int t = 0; t < 4; ++t) {
    const int nn = n0 + t * 16 + n;
    const float bv = bias[nn];
#pragma unroll
    for (int r = 0; r < 8; ++r) {
      const int m = m0 + r + mbase;
      float v = acc[t][r] + bv;
      if (mode == 2) {
        ((float*)outp)[(size_t)m * HID + nn] = v;
      } else {
        const int bidx = m / SEQ, s = m % SEQ;
        const int head = nn / HDIM, h = nn % HDIM;
        size_t idx = (mode == 0)
            ? ((((size_t)bidx * NHEAD + head) * SEQ + s) * HDIM + h)
            : ((((size_t)bidx * NHEAD + head) * HDIM + h) * SEQ + s);
        ((__bf16*)outp)[idx] = f32_to_bf16(v);
      }
    }
  }
}

// ---------------------------------------------------------------------------
// Flash attention: per (b, head, 64 query rows) block; 4 waves x 16 rows.
// Q,K in [B,NH,S,HD] bf16; V transposed [B,NH,HD,S] bf16; mask int32 [B,S,S].
// Streams 32 keys per iteration with online softmax; writes ctx bf16 [B,S,H].
// ---------------------------------------------------------------------------
__global__ __launch_bounds__(128) void attn_kernel(
    const __bf16* __restrict__ Qp, const __bf16* __restrict__ Kp,
    const __bf16* __restrict__ Vt, const int* __restrict__ mask,
    __bf16* __restrict__ ctx) {
  __shared__ __bf16 lds[4 * 16 * 32];   // per-wave 16x32 P staging tile

  const int wave   = threadIdx.x >> 5;
  const int lane   = threadIdx.x & 31;
  const int qtiles = SEQ / 64;                 // 32
  const int qt     = blockIdx.x % qtiles;
  const int bh     = blockIdx.x / qtiles;      // b*NHEAD + head
  const int head   = bh % NHEAD;
  const int b      = bh / NHEAD;

  const int q0 = qt * 64 + wave * 16;
  const __bf16* Qh = Qp + (size_t)bh * SEQ * HDIM;
  const __bf16* Kh = Kp + (size_t)bh * SEQ * HDIM;
  const __bf16* Vh = Vt + (size_t)bh * HDIM * SEQ;
  const int* maskb = mask + (size_t)b * SEQ * SEQ;

  const int row   = lane & 15;
  const int kbase = (lane < 16) ? 0 : 8;
  const int n     = lane & 15;
  const int khalf = (lane < 16) ? 0 : 16;
  const int mbase = (lane < 16) ? 0 : 8;

  // Q tile (16x64) as two 16x32 A fragments, loaded once
  const __bf16* qrow = Qh + (size_t)(q0 + row) * HDIM;
  v16bf aQ0 = *(const v16bf*)(qrow + khalf);
  v16bf aQ1 = *(const v16bf*)(qrow + 32 + khalf);

  v8f acc0 = {}, acc1 = {}, acc2 = {}, acc3 = {};
  float mrun[8], lrun[8];
#pragma unroll
  for (int r = 0; r < 8; ++r) { mrun[r] = -3.0e38f; lrun[r] = 0.0f; }

  __bf16* pl = lds + wave * (16 * 32);

  for (int kb = 0; kb < SEQ; kb += 32) {
    // S(16x32) = Q(16x64) @ K^T : two 16x16 column tiles, K-dim split 0..31/32..63
    const __bf16* kr0 = Kh + (size_t)(kb + n) * HDIM + khalf;
    const __bf16* kr1 = Kh + (size_t)(kb + 16 + n) * HDIM + khalf;
    v8f z = {};
    v8f s0 = wmma_bf16(aQ0, *(const v16bf*)(kr0), z);
    s0     = wmma_bf16(aQ1, *(const v16bf*)(kr0 + 32), s0);
    v8f s1 = wmma_bf16(aQ0, *(const v16bf*)(kr1), z);
    s1     = wmma_bf16(aQ1, *(const v16bf*)(kr1 + 32), s1);

    // online softmax (scale = 1/sqrt(64) = 0.125, mask==0 -> -1e9)
#pragma unroll
    for (int r = 0; r < 8; ++r) {
      const int* mr = maskb + (size_t)(q0 + r + mbase) * SEQ + kb;
      float a = s0[r] * 0.125f;
      float c = s1[r] * 0.125f;
      if (mr[n] == 0)      a = -1.0e9f;
      if (mr[16 + n] == 0) c = -1.0e9f;

      float t = fmaxf(a, c);                 // row max across 16 lanes (half-wave)
      t = fmaxf(t, __shfl_xor(t, 1));
      t = fmaxf(t, __shfl_xor(t, 2));
      t = fmaxf(t, __shfl_xor(t, 4));
      t = fmaxf(t, __shfl_xor(t, 8));
      float mnew  = fmaxf(mrun[r], t);
      float alpha = __expf(mrun[r] - mnew);
      float p0 = __expf(a - mnew);
      float p1 = __expf(c - mnew);
      float ps = p0 + p1;                    // row sum across 16 lanes
      ps += __shfl_xor(ps, 1);
      ps += __shfl_xor(ps, 2);
      ps += __shfl_xor(ps, 4);
      ps += __shfl_xor(ps, 8);
      lrun[r] = lrun[r] * alpha + ps;
      mrun[r] = mnew;
      acc0[r] *= alpha; acc1[r] *= alpha; acc2[r] *= alpha; acc3[r] *= alpha;
      // stage P in C-layout -> row-major LDS tile (same-wave LDS is in-order)
      pl[(r + mbase) * 32 + n]      = f32_to_bf16(p0);
      pl[(r + mbase) * 32 + 16 + n] = f32_to_bf16(p1);
    }

    // reload P as a 16x32 A fragment
    const __bf16* pr = pl + row * 32;
    v16bf aP = *(const v16bf*)(pr + khalf);

    // O(16x64) += P(16x32) @ V(32x64); V transposed so loads are contiguous
    const __bf16* v0 = Vh + (size_t)(0  + n) * SEQ + kb + khalf;
    const __bf16* v1 = Vh + (size_t)(16 + n) * SEQ + kb + khalf;
    const __bf16* v2 = Vh + (size_t)(32 + n) * SEQ + kb + khalf;
    const __bf16* v3 = Vh + (size_t)(48 + n) * SEQ + kb + khalf;
    acc0 = wmma_bf16(aP, *(const v16bf*)v0, acc0);
    acc1 = wmma_bf16(aP, *(const v16bf*)v1, acc1);
    acc2 = wmma_bf16(aP, *(const v16bf*)v2, acc2);
    acc3 = wmma_bf16(aP, *(const v16bf*)v3, acc3);
  }

  // normalize and emit context in [B, S, H] bf16 for the output projection
#pragma unroll
  for (int r = 0; r < 8; ++r) {
    float inv = 1.0f / lrun[r];
    __bf16* crow = ctx + ((size_t)b * SEQ + q0 + r + mbase) * HID + head * HDIM;
    crow[ 0 + n] = f32_to_bf16(acc0[r] * inv);
    crow[16 + n] = f32_to_bf16(acc1[r] * inv);
    crow[32 + n] = f32_to_bf16(acc2[r] * inv);
    crow[48 + n] = f32_to_bf16(acc3[r] * inv);
  }
}

// ---------------------------------------------------------------------------
// Launcher
// ---------------------------------------------------------------------------
extern "C" void kernel_launch(void* const* d_in, const int* in_sizes, int n_in,
                              void* d_out, int out_size, void* d_ws, size_t ws_size,
                              hipStream_t stream) {
  const float* q    = (const float*)d_in[0];
  const float* k    = (const float*)d_in[1];
  const float* v    = (const float*)d_in[2];
  const int*   mask = (const int*)d_in[3];
  const float* Wq = (const float*)d_in[4];  const float* bq = (const float*)d_in[5];
  const float* Wk = (const float*)d_in[6];  const float* bk = (const float*)d_in[7];
  const float* Wv = (const float*)d_in[8];  const float* bv = (const float*)d_in[9];
  const float* Wo = (const float*)d_in[10]; const float* bo = (const float*)d_in[11];

  const size_t SA = (size_t)MROWS * HID;   // 4,194,304 elements per activation
  const size_t SW = (size_t)HID * HID;     //   262,144 elements per weight

  __bf16* qb  = (__bf16*)d_ws;             // ws layout (~61 MB total, bf16)
  __bf16* kb  = qb  + SA;
  __bf16* vb  = kb  + SA;
  __bf16* wqb = vb  + SA;
  __bf16* wkb = wqb + SW;
  __bf16* wvb = wkb + SW;
  __bf16* wob = wvb + SW;
  __bf16* Qp  = wob + SW;                  // [B,NH,S,HD]
  __bf16* Kp  = Qp  + SA;                  // [B,NH,S,HD]
  __bf16* Vt  = Kp  + SA;                  // [B,NH,HD,S]
  __bf16* ctx = Vt  + SA;                  // [B,S,H]

  dim3 cb(256), cg(1024);
  cvt_f32_bf16<<<cg, cb, 0, stream>>>(q,  qb,  (int)SA);
  cvt_f32_bf16<<<cg, cb, 0, stream>>>(k,  kb,  (int)SA);
  cvt_f32_bf16<<<cg, cb, 0, stream>>>(v,  vb,  (int)SA);
  cvt_f32_bf16<<<cg, cb, 0, stream>>>(Wq, wqb, (int)SW);
  cvt_f32_bf16<<<cg, cb, 0, stream>>>(Wk, wkb, (int)SW);
  cvt_f32_bf16<<<cg, cb, 0, stream>>>(Wv, wvb, (int)SW);
  cvt_f32_bf16<<<cg, cb, 0, stream>>>(Wo, wob, (int)SW);

  dim3 gb(128), gg(MROWS / 64, HID / 64);
  gemm_xwT_kernel<<<gg, gb, 0, stream>>>(qb, wqb, bq, (void*)Qp, 0);
  gemm_xwT_kernel<<<gg, gb, 0, stream>>>(kb, wkb, bk, (void*)Kp, 0);
  gemm_xwT_kernel<<<gg, gb, 0, stream>>>(vb, wvb, bv, (void*)Vt, 1);

  attn_kernel<<<dim3(BATCH * NHEAD * (SEQ / 64)), dim3(128), 0, stream>>>(
      Qp, Kp, Vt, mask, ctx);

  gemm_xwT_kernel<<<gg, gb, 0, stream>>>(ctx, wob, bo, d_out, 2);
}
